// ITLRegularizer_59545426592220
// MI455X (gfx1250) — matrix operand order repaired and checked by reading, not compiled, not run on hardware
//
#include <hip/hip_runtime.h>
#include <math.h>
#include <stdint.h>

// ---------------------------------------------------------------------------
// ITL Parzen divergence, fused for MI455X (gfx1250, wave32, WMMA):
//   G(A,B)[i][j] = exp(-(||a_i||^2 + ||b_j||^2 - 2 a_i.b_j))
//   out = (sum Gxx + sum Gzz - 2 sum Gxz) / (N^2 * sqrt(2*pi))
// Strategy: bf16 hi/lo split GEMM (3x v_wmma_f32_16x16x32_bf16 per K-step)
// with exact f32 row norms, async global->LDS double-buffered B staging,
// fused exp + deterministic tree reductions.
// ---------------------------------------------------------------------------

typedef __attribute__((ext_vector_type(16))) __bf16        v16bf;
typedef __attribute__((ext_vector_type(8)))  float         v8f;
typedef __attribute__((ext_vector_type(4)))  unsigned int  u32x4;
typedef __attribute__((ext_vector_type(4)))  int           v4i;

union Frag { v16bf bf; u32x4 u4[2]; };

#define DIM  256   // feature dimension D
#define TILE 128   // output tile per workgroup
#define BPAD 40    // LDS B-row pitch in ushorts (80B) -> conflict-free b128 reads

#define AS1 __attribute__((address_space(1)))
#define AS3 __attribute__((address_space(3)))

// CDNA5 async global->LDS copy (ASYNCcnt-tracked); sync fallback otherwise.
#if __has_builtin(__builtin_amdgcn_global_load_async_to_lds_b128)
__device__ __forceinline__ void async_cp_b128(const void* g, void* l) {
  // Builtin wants typed AS-qualified pointers; go via integers (generic LDS
  // pointers carry the LDS byte offset in addr[31:0] per the aperture rules).
  __builtin_amdgcn_global_load_async_to_lds_b128(
      (AS1 v4i*)(unsigned long long)(uintptr_t)g,
      (AS3 v4i*)(unsigned)(uintptr_t)l, 0, 0);
}
#if __has_builtin(__builtin_amdgcn_s_wait_asynccnt)
#define ASYNC_WAIT() __builtin_amdgcn_s_wait_asynccnt(0)
#else
#define ASYNC_WAIT() asm volatile("s_wait_asynccnt 0" ::: "memory")
#endif
#else
__device__ __forceinline__ void async_cp_b128(const void* g, void* l) {
  *(u32x4*)l = *(const u32x4*)g;
}
#define ASYNC_WAIT()
#endif

__device__ __forceinline__ unsigned short f32_bf16_rne(float f) {
  unsigned u = __float_as_uint(f);
  unsigned r = u + 0x7FFFu + ((u >> 16) & 1u);   // round to nearest even
  return (unsigned short)(r >> 16);
}

// Split each f32 into bf16 hi + bf16 lo planes; exact f32 row sum-of-squares.
__global__ __launch_bounds__(DIM) void itl_prep(const float* __restrict__ src,
                                                unsigned short* __restrict__ hi,
                                                unsigned short* __restrict__ lo,
                                                float* __restrict__ nrm) {
  const int row = blockIdx.x;
  const int col = threadIdx.x;
  const int idx = row * DIM + col;
  const float x = src[idx];

  unsigned u  = __float_as_uint(x);
  unsigned rb = (u + 0x7FFFu + ((u >> 16) & 1u)) & 0xFFFF0000u;
  const float fh = __uint_as_float(rb);          // hi as f32 (bf16-exact)
  const float fl = x - fh;                       // exact residual
  hi[idx] = (unsigned short)(rb >> 16);
  lo[idx] = f32_bf16_rne(fl);

  __shared__ float red[DIM];
  red[col] = x * x;
  __syncthreads();
  for (int s = DIM / 2; s > 0; s >>= 1) {
    if (col < s) red[col] += red[col + s];
    __syncthreads();
  }
  if (col == 0) nrm[row] = red[0];
}

// Stage one 128x32 B tile (hi+lo planes) for K-step kk into LDS buffer.
// 1024 x 16B chunks split across 256 threads (4 async b128 copies each).
__device__ __forceinline__ void stage_B(unsigned short* sBbuf,
                                        const unsigned short* __restrict__ Bhi,
                                        const unsigned short* __restrict__ Blo,
                                        int colTile, int kk, int tid) {
#pragma unroll
  for (int j = 0; j < 4; ++j) {
    const int c = (j << 8) | tid;        // 0..1023
    const int p = c >> 9;                // plane: 0=hi 1=lo
    const int r = (c >> 2) & 127;        // B row within tile
    const int s = (c & 3) << 3;          // element offset 0/8/16/24
    const unsigned short* g = (p ? Blo : Bhi) + (size_t)(colTile + r) * DIM + kk + s;
    unsigned short* l = sBbuf + p * (128 * BPAD) + r * BPAD + s;
    async_cp_b128(g, l);
  }
}

// 128x128 Gram tile per block; 8 waves (4 along M x 2 along N), each wave
// computing a 32x64 region as 2x4 WMMA subtiles. hi*hi + hi*lo + lo*hi.
__global__ __launch_bounds__(256) void itl_gram(
    const unsigned short* __restrict__ Ahi, const unsigned short* __restrict__ Alo,
    const float* __restrict__ na,
    const unsigned short* __restrict__ Bhi, const unsigned short* __restrict__ Blo,
    const float* __restrict__ nb,
    float* __restrict__ partial) {
  __shared__ __attribute__((aligned(16))) unsigned short sB[2][2][128][BPAD];

  const int tid  = threadIdx.x;
  const int wave = tid >> 5;
  const int lane = tid & 31;
  const int lm   = lane & 15;
  const int kA   = (lane >> 4) << 3;   // A-frag K base per lane half: 0 / 8
  const int kB   = (lane >> 4) << 3;   // B-frag K base per lane half: 0 / 8

  const int colTile = blockIdx.x * TILE;
  const int cw      = (wave >> 2) * 64;              // local col base of wave
  const int waveR   = blockIdx.y * TILE + (wave & 3) * 32;
  const int waveC   = colTile + cw;

  const v8f vzero = {0.f, 0.f, 0.f, 0.f, 0.f, 0.f, 0.f, 0.f};
  v8f acc[2][4];
#pragma unroll
  for (int mi = 0; mi < 2; ++mi)
#pragma unroll
    for (int ni = 0; ni < 4; ++ni) acc[mi][ni] = vzero;

  stage_B(&sB[0][0][0][0], Bhi, Blo, colTile, 0, tid);   // prologue, buffer 0

  int buf = 0;
  for (int kk = 0; kk < DIM; kk += 32) {
    ASYNC_WAIT();            // this thread's async writes to sB[buf] done
    __syncthreads();         // everyone's writes visible; prior reads retired
    if (kk + 32 < DIM)
      stage_B(&sB[buf ^ 1][0][0][0], Bhi, Blo, colTile, kk + 32, tid);

    // A fragments (16x32 bf16) from global: lane lm = row; chunks at kA, kA+16.
    Frag ah[2], al[2];
#pragma unroll
    for (int mi = 0; mi < 2; ++mi) {
      const size_t abase = (size_t)(waveR + mi * 16 + lm) * DIM + kk + kA;
      ah[mi].u4[0] = *(const u32x4*)(Ahi + abase);
      ah[mi].u4[1] = *(const u32x4*)(Ahi + abase + 16);
      al[mi].u4[0] = *(const u32x4*)(Alo + abase);
      al[mi].u4[1] = *(const u32x4*)(Alo + abase + 16);
      if (kk + 32 < DIM) {                       // global_prefetch_b8 (WGP)
        __builtin_prefetch(Ahi + abase + 32, 0, 3);
        __builtin_prefetch(Alo + abase + 32, 0, 3);
      }
    }

    const unsigned short* sb0 = &sB[buf][0][0][0];
    const unsigned short* sb1 = &sB[buf][1][0][0];
#pragma unroll
    for (int ni = 0; ni < 4; ++ni) {
      const int lbase = (cw + ni * 16 + lm) * BPAD + kB;
      Frag bh, bl;
      bh.u4[0] = *(const u32x4*)(sb0 + lbase);
      bh.u4[1] = *(const u32x4*)(sb0 + lbase + 16);
      bl.u4[0] = *(const u32x4*)(sb1 + lbase);
      bl.u4[1] = *(const u32x4*)(sb1 + lbase + 16);
#pragma unroll
      for (int mi = 0; mi < 2; ++mi) {
        acc[mi][ni] = __builtin_amdgcn_wmma_f32_16x16x32_bf16(
            false, ah[mi].bf, false, bh.bf, (short)0, acc[mi][ni], false, false);
        acc[mi][ni] = __builtin_amdgcn_wmma_f32_16x16x32_bf16(
            false, ah[mi].bf, false, bl.bf, (short)0, acc[mi][ni], false, false);
        acc[mi][ni] = __builtin_amdgcn_wmma_f32_16x16x32_bf16(
            false, al[mi].bf, false, bh.bf, (short)0, acc[mi][ni], false, false);
      }
    }
    buf ^= 1;
  }

  // Fused epilogue: d2 = ||a||^2 + ||b||^2 - 2*dot ; sum exp(-d2).
  // C/D layout: VGPR r -> M = r + (lane>=16 ? 8 : 0), N = lane & 15.
  float local = 0.f;
#pragma unroll
  for (int mi = 0; mi < 2; ++mi) {
    const int rbase = waveR + mi * 16 + kA;     // kA==8 matches hi-lane M offset
    float nav[8];
#pragma unroll
    for (int r = 0; r < 8; ++r) nav[r] = na[rbase + r];
#pragma unroll
    for (int ni = 0; ni < 4; ++ni) {
      const float nbv = nb[waveC + ni * 16 + lm];
#pragma unroll
      for (int r = 0; r < 8; ++r) {
        const float d2 = nav[r] + nbv - 2.0f * acc[mi][ni][r];
        local += __expf(-d2);
      }
    }
  }

  __shared__ float red[256];
  red[tid] = local;
  __syncthreads();
  for (int s = 128; s > 0; s >>= 1) {
    if (tid < s) red[tid] += red[tid + s];
    __syncthreads();
  }
  if (tid == 0) partial[blockIdx.y * gridDim.x + blockIdx.x] = red[0];
}

// Deterministic final combine: out = (Sxx + Szz - 2*Sxz) * scale.
__global__ __launch_bounds__(256) void itl_finalize(const float* __restrict__ p,
                                                    int nb, float scale,
                                                    float* __restrict__ out) {
  const int tid = threadIdx.x;
  float s = 0.f;
  for (int i = tid; i < nb; i += 256)
    s += p[i] + p[nb + i] - 2.0f * p[2 * nb + i];
  __shared__ float red[256];
  red[tid] = s;
  __syncthreads();
  for (int k = 128; k > 0; k >>= 1) {
    if (tid < k) red[tid] += red[tid + k];
    __syncthreads();
  }
  if (tid == 0) out[0] = red[0] * scale;
}

extern "C" void kernel_launch(void* const* d_in, const int* in_sizes, int n_in,
                              void* d_out, int out_size, void* d_ws, size_t ws_size,
                              hipStream_t stream) {
  const float* X = (const float*)d_in[0];
  const float* Z = (const float*)d_in[1];
  const int n = in_sizes[0] / DIM;              // 8192
  const size_t nd = (size_t)n * DIM;

  // Workspace layout: 4 bf16 planes (16 MB), 2 norm vectors, 3*nb partials.
  unsigned short* Xhi = (unsigned short*)d_ws;
  unsigned short* Xlo = Xhi + nd;
  unsigned short* Zhi = Xlo + nd;
  unsigned short* Zlo = Zhi + nd;
  float* aa       = (float*)(Zlo + nd);
  float* bbv      = aa + n;
  float* partials = bbv + n;

  itl_prep<<<n, DIM, 0, stream>>>(X, Xhi, Xlo, aa);
  itl_prep<<<n, DIM, 0, stream>>>(Z, Zhi, Zlo, bbv);

  const int nt = n / TILE;
  const int nb = nt * nt;
  dim3 grid(nt, nt);
  itl_gram<<<grid, 256, 0, stream>>>(Xhi, Xlo, aa, Xhi, Xlo, aa, partials);
  itl_gram<<<grid, 256, 0, stream>>>(Zhi, Zlo, bbv, Zhi, Zlo, bbv, partials + nb);
  itl_gram<<<grid, 256, 0, stream>>>(Xhi, Xlo, aa, Zhi, Zlo, bbv, partials + 2 * nb);

  const double inv = 1.0 / ((double)n * (double)n * sqrt(2.0 * 3.14159265358979323846));
  itl_finalize<<<1, 256, 0, stream>>>(partials, nb, (float)inv, (float*)d_out);
}
